// LSTMSeq2Seq_67216238182500
// MI455X (gfx1250) — matrix operand order repaired and checked by reading, not compiled, not run on hardware
//
#include <hip/hip_runtime.h>

// ============================================================================
// LSTM seq2seq for MI455X (gfx1250, wave32, WMMA, LDS-staged weights).
// bf16 operands + f32 accumulation via v_wmma_f32_16x16x32_bf16.
// Weights pre-transposed to [N][K] bf16 so fragments are contiguous loads.
// Step kernel: block = 8 waves sharing one 16-col gate-tile set; weights for
// each 32-K slice staged into LDS once per block (8x cache-traffic cut);
// each wave owns 32 rows -> 8 f32 accumulators (2 row tiles x 4 gates).
// ============================================================================

#define DEV __device__ __forceinline__

typedef __bf16 bf16;
typedef __attribute__((ext_vector_type(16))) __bf16 v16bf;
typedef __attribute__((ext_vector_type(8)))  __bf16 v8bf;
typedef __attribute__((ext_vector_type(8)))  float  v8f;

static constexpr int  B_    = 512;
static constexpr int  TPAST = 128;
static constexpr int  TFUT  = 48;
static constexpr int  FIN   = 64;
static constexpr int  FOUT  = 16;
static constexpr int  H     = 512;
static constexpr int  G4H   = 4 * H;            // 2048
static constexpr long MD    = (long)B_ * TFUT;  // 24576 decoder rows

// ---------------------------------------------------------------------------
// helpers
// ---------------------------------------------------------------------------
DEV int lane_id() { return threadIdx.x & 31; }

DEV float sigm(float x) { return 1.0f / (1.0f + __expf(-x)); }

DEV float tanh_(float x) {
  float e = __expf(2.0f * x);
  float t = (e - 1.0f) / (e + 1.0f);
  return (x > 10.0f) ? 1.0f : ((x < -10.0f) ? -1.0f : t);
}

// A-fragment: 16x32 bf16 tile, row-major source with row stride `stride`.
// lanes 0-15 hold M=lane, elems 0..15 = K kb..kb+15;
// lanes 16-31 hold K kb+16..kb+31 (contiguous, matches load_b).
DEV v16bf load_a(const bf16* __restrict__ base, long stride, int m0, int kb) {
  int lane = lane_id();
  int half = lane >> 4;
  const bf16* p = base + (long)(m0 + (lane & 15)) * stride + kb + half * 16;
  v8bf lo = *(const v8bf*)p;         // K = kb + half*16 .. +7
  v8bf hi = *(const v8bf*)(p + 8);   // K = kb + half*16 + 8 .. +7
  v16bf r;
#pragma unroll
  for (int i = 0; i < 8; ++i) { r[i] = lo[i]; r[i + 8] = hi[i]; }
  return r;
}

// B-fragment: 32x16 bf16 tile from a TRANSPOSED weight matrix Wt[N][K]
// (ldk = K). Lane holds column n0+(lane&15); elems 0..15 are the 16
// consecutive K values starting at kb + (lane<16 ? 0 : 16).
DEV v16bf load_b(const bf16* __restrict__ Wt, int ldk, int n0, int kb) {
  int lane = lane_id();
  int half = lane >> 4;
  const bf16* p = Wt + (long)(n0 + (lane & 15)) * ldk + kb + half * 16;
  v8bf lo = *(const v8bf*)p;
  v8bf hi = *(const v8bf*)(p + 8);
  v16bf r;
#pragma unroll
  for (int i = 0; i < 8; ++i) { r[i] = lo[i]; r[i + 8] = hi[i]; }
  return r;
}

// B-fragment from LDS stage: sB[g][col][k] with col in 0..15, k in 0..31.
DEV v16bf load_b_lds(const bf16* sB, int g) {
  int lane = lane_id();
  int half = lane >> 4;
  const bf16* p = sB + ((g * 16 + (lane & 15)) * 32) + half * 16;
  v8bf lo = *(const v8bf*)p;
  v8bf hi = *(const v8bf*)(p + 8);
  v16bf r;
#pragma unroll
  for (int i = 0; i < 8; ++i) { r[i] = lo[i]; r[i + 8] = hi[i]; }
  return r;
}

DEV v8f wmma_bf16(v16bf a, v16bf b, v8f c) {
  // (neg_a, A, neg_b, B, c_mod, C, reuse_a, reuse_b)
  return __builtin_amdgcn_wmma_f32_16x16x32_bf16(false, a, false, b,
                                                 (short)0, c, false, false);
}

// ---------------------------------------------------------------------------
// fused LSTM step:  z = [x_t | h_prev] @ Wt^T + b ; gates ; write h, c
// Block: 8 waves, one shared column tile tn (4 gate tiles staged in LDS),
// wave w owns rows [rg*256 + w*32, +32).  Requires rows % 256 == 0.
// Also used for decoder layer 2 (Dx == Ktot, hprev unused, c in-place).
// ---------------------------------------------------------------------------
__global__ void k_lstm_step(const bf16* __restrict__ xbase, long xstride, int Dx,
                            const bf16* __restrict__ hprev,
                            const bf16* __restrict__ Wt, int Ktot,
                            const float* __restrict__ bias,
                            float* __restrict__ cstate,   // [rows][H] in/out
                            bf16* __restrict__ hout,      // [rows][H]
                            bf16* __restrict__ hseq,      // optional [rows][H]
                            int rows) {
  __shared__ alignas(16) bf16 sB[4 * 16 * 32];   // 4 KB: 4 gates x 16 cols x 32 K

  (void)rows;
  int tid  = threadIdx.x;
  int wave = tid >> 5;
  int cn   = blockIdx.x & 31;         // 32 column tiles (H/16)
  int rg   = blockIdx.x >> 5;         // row group of 256 rows
  int tn   = cn << 4;
  int tm   = rg * 256 + wave * 32;

  // per-thread weight staging slot: one 16B chunk per thread per K-slice
  // thread -> (gate = tid>>6, col = (tid>>2)&15, seg = tid&3)
  const bf16* wsrc = Wt
      + (long)(tn + (tid >> 6) * H + ((tid >> 2) & 15)) * Ktot + (tid & 3) * 8;
  bf16* sdst = sB + tid * 8;          // contiguous: tid*16 bytes

  v8f acc[2][4] = {};

  for (int kb = 0; kb < Ktot; kb += 32) {
    // issue weight fetch for this slice (latency overlaps the barrier)
    v8bf w = *(const v8bf*)(wsrc + kb);
    __builtin_prefetch(wsrc + kb + 32, 0, 3);
    __syncthreads();                  // readers of previous slice are done
    *(v8bf*)sdst = w;

    // A fragments (per-wave rows; uniform source select, x-part then h-part)
    v16bf a0, a1;
    if (kb < Dx) {
      a0 = load_a(xbase, xstride, tm, kb);
      a1 = load_a(xbase, xstride, tm + 16, kb);
    } else {
      a0 = load_a(hprev, H, tm, kb - Dx);
      a1 = load_a(hprev, H, tm + 16, kb - Dx);
    }
    __syncthreads();                  // stage visible to all waves

#pragma unroll
    for (int g = 0; g < 4; ++g) {
      v16bf b = load_b_lds(sB, g);
      acc[0][g] = wmma_bf16(a0, b, acc[0][g]);
      acc[1][g] = wmma_bf16(a1, b, acc[1][g]);
    }
  }

  int lane = lane_id();
  int half = lane >> 4;
  int nc = tn + (lane & 15);
  float bi = bias[0 * H + nc], bf_ = bias[1 * H + nc];
  float bg = bias[2 * H + nc], bo  = bias[3 * H + nc];
#pragma unroll
  for (int rt = 0; rt < 2; ++rt) {
#pragma unroll
    for (int r = 0; r < 8; ++r) {
      int row = tm + rt * 16 + r + half * 8;
      float zi = acc[rt][0][r] + bi, zf = acc[rt][1][r] + bf_;
      float zg = acc[rt][2][r] + bg, zo = acc[rt][3][r] + bo;
      long idx = (long)row * H + nc;
      float cp = cstate[idx];
      float c  = sigm(zf) * cp + sigm(zi) * tanh_(zg);
      float h  = sigm(zo) * tanh_(c);
      cstate[idx] = c;
      hout[idx] = (bf16)h;
      if (hseq) hseq[idx] = (bf16)h;
    }
  }
}

// ---------------------------------------------------------------------------
// r2 = h2 @ dec1_Wr  (plain GEMM, f32 out)   M=512, N=2048, K=512
// ---------------------------------------------------------------------------
__global__ void k_r2(const bf16* __restrict__ h2, const bf16* __restrict__ Wrt,
                     float* __restrict__ r2) {
  int wave = threadIdx.x >> 5;
  int tile = blockIdx.x * 8 + wave;        // (512/32) x 128 tiles
  int tm = (tile >> 7) << 5;               // 32 rows per wave
  int tn = (tile & 127) << 4;
  v8f acc0 = {}, acc1 = {};
  for (int kb = 0; kb < H; kb += 32) {
    v16bf a0 = load_a(h2, H, tm, kb);
    v16bf a1 = load_a(h2, H, tm + 16, kb);
    v16bf b  = load_b(Wrt, H, tn, kb);
    acc0 = wmma_bf16(a0, b, acc0);
    acc1 = wmma_bf16(a1, b, acc1);
  }
  int lane = lane_id(), half = lane >> 4, nc = tn + (lane & 15);
#pragma unroll
  for (int r = 0; r < 8; ++r) {
    r2[(long)(tm + r + half * 8) * G4H + nc]      = acc0[r];
    r2[(long)(tm + 16 + r + half * 8) * G4H + nc] = acc1[r];
  }
}

// ---------------------------------------------------------------------------
// decoder layer 1: z1 = target @ dec1_Wk (K padded to 32) + r2[b] + b ; gates
// with c_prev = c2[b] broadcast over t.  rows = B*TFUT.
// ---------------------------------------------------------------------------
__global__ void k_dec1(const bf16* __restrict__ tgt,   // [MD][32]
                       const bf16* __restrict__ Wkt,   // [2048][32]
                       const float* __restrict__ r2,   // [B][4H]
                       const float* __restrict__ c2,   // [B][H]
                       const float* __restrict__ bias, // [4H]
                       bf16* __restrict__ h1d,         // [MD][H]
                       float* __restrict__ c1d) {      // [MD][H]
  int wave = threadIdx.x >> 5;
  int tile = blockIdx.x * 8 + wave;        // 1536 x 32 tiles
  int tm = (tile >> 5) << 4;
  int tn = (tile & 31) << 4;
  v8f acc[4] = {};
  v16bf a = load_a(tgt, 32, tm, 0);
#pragma unroll
  for (int g = 0; g < 4; ++g) {
    v16bf b = load_b(Wkt, 32, tn + g * H, 0);
    acc[g] = wmma_bf16(a, b, acc[g]);
  }
  int lane = lane_id(), half = lane >> 4, nc = tn + (lane & 15);
  float bi = bias[nc], bf_ = bias[H + nc];
  float bg = bias[2 * H + nc], bo = bias[3 * H + nc];
#pragma unroll
  for (int r = 0; r < 8; ++r) {
    int row = tm + r + half * 8;
    int bb = row / TFUT;
    float zi = acc[0][r] + bi  + r2[(long)bb * G4H + 0 * H + nc];
    float zf = acc[1][r] + bf_ + r2[(long)bb * G4H + 1 * H + nc];
    float zg = acc[2][r] + bg  + r2[(long)bb * G4H + 2 * H + nc];
    float zo = acc[3][r] + bo  + r2[(long)bb * G4H + 3 * H + nc];
    float cp = c2[(long)bb * H + nc];
    float c  = sigm(zf) * cp + sigm(zi) * tanh_(zg);
    float h  = sigm(zo) * tanh_(c);
    h1d[(long)row * H + nc] = (bf16)h;
    c1d[(long)row * H + nc] = c;
  }
}

// ---------------------------------------------------------------------------
// out = h2d @ fc_W + fc_b   M=24576, N=16, K=512  (f32 output)
// ---------------------------------------------------------------------------
__global__ void k_fc(const bf16* __restrict__ h2d,
                     const bf16* __restrict__ Wt,      // [16][512]
                     const float* __restrict__ bias,   // [16]
                     float* __restrict__ out) {
  int wave = threadIdx.x >> 5;
  int tm = (blockIdx.x * 8 + wave) << 5;   // 32 rows per wave
  v8f acc0 = {}, acc1 = {};
  for (int kb = 0; kb < H; kb += 32) {
    v16bf a0 = load_a(h2d, H, tm, kb);
    v16bf a1 = load_a(h2d, H, tm + 16, kb);
    v16bf b  = load_b(Wt, H, 0, kb);
    acc0 = wmma_bf16(a0, b, acc0);
    acc1 = wmma_bf16(a1, b, acc1);
  }
  int lane = lane_id(), half = lane >> 4, nc = lane & 15;
  float bb = bias[nc];
#pragma unroll
  for (int r = 0; r < 8; ++r) {
    out[(long)(tm + r + half * 8) * FOUT + nc]      = acc0[r] + bb;
    out[(long)(tm + 16 + r + half * 8) * FOUT + nc] = acc1[r] + bb;
  }
}

// ---------------------------------------------------------------------------
// prep kernels
// ---------------------------------------------------------------------------
__global__ void k_cvt_f32_bf16(const float* __restrict__ s, bf16* __restrict__ d,
                               long n) {
  long i = (long)blockIdx.x * blockDim.x + threadIdx.x;
  if (i < n) d[i] = (bf16)s[i];
}

__global__ void k_prep_tgt(const float* __restrict__ t, bf16* __restrict__ d) {
  long i = (long)blockIdx.x * blockDim.x + threadIdx.x;   // over MD*32
  if (i >= MD * 32) return;
  int f = (int)(i & 31);
  long m = i >> 5;
  d[i] = (f < FOUT) ? (bf16)t[m * FOUT + f] : (bf16)0.0f;
}

// dst[n*(Ka+Kb)+k] = (k<Ka ? A[k][n] : Bs[k-Ka][n]), sources [K][N] row-major.
__global__ void k_prep_cat(const float* __restrict__ A, int Ka,
                           const float* __restrict__ Bs, int Kb,
                           int N, bf16* __restrict__ dst) {
  int K = Ka + Kb;
  long total = (long)N * K;
  long i = (long)blockIdx.x * blockDim.x + threadIdx.x;
  if (i >= total) return;
  int k = (int)(i % K);
  long n = i / K;
  float v = (k < Ka) ? A[(long)k * N + n] : Bs[(long)(k - Ka) * N + n];
  dst[i] = (bf16)v;
}

__global__ void k_prep_sum(const float* __restrict__ A,
                           const float* __restrict__ Bs,
                           int K, int N, bf16* __restrict__ dst) {
  long total = (long)N * K;
  long i = (long)blockIdx.x * blockDim.x + threadIdx.x;
  if (i >= total) return;
  int k = (int)(i % K);
  long n = i / K;
  dst[i] = (bf16)(A[(long)k * N + n] + Bs[(long)k * N + n]);
}

__global__ void k_prep_pad(const float* __restrict__ Wk, bf16* __restrict__ dst) {
  // dst [2048][32] <- transpose of [16][2048], K padded 16->32 with zeros
  long i = (long)blockIdx.x * blockDim.x + threadIdx.x;
  if (i >= 2048L * 32) return;
  int k = (int)(i & 31);
  long n = i >> 5;
  dst[i] = (k < FOUT) ? (bf16)Wk[(long)k * G4H + n] : (bf16)0.0f;
}

__global__ void k_zero_f32(float* p, long n) {
  long i = (long)blockIdx.x * blockDim.x + threadIdx.x;
  if (i < n) p[i] = 0.0f;
}
__global__ void k_zero_bf16(bf16* p, long n) {
  long i = (long)blockIdx.x * blockDim.x + threadIdx.x;
  if (i < n) p[i] = (bf16)0.0f;
}

// ---------------------------------------------------------------------------
// host launcher
// ---------------------------------------------------------------------------
static inline long ceil_div(long a, long b) { return (a + b - 1) / b; }

extern "C" void kernel_launch(void* const* d_in, const int* in_sizes, int n_in,
                              void* d_out, int out_size, void* d_ws,
                              size_t ws_size, hipStream_t stream) {
  (void)in_sizes; (void)n_in; (void)out_size; (void)ws_size;

  const float* inp      = (const float*)d_in[0];
  const float* tgt      = (const float*)d_in[1];
  const float* enc1_Wk  = (const float*)d_in[2];
  const float* enc1_Wr  = (const float*)d_in[3];
  const float* enc1_b   = (const float*)d_in[4];
  const float* enc2_Wk  = (const float*)d_in[5];
  const float* enc2_Wr  = (const float*)d_in[6];
  const float* enc2_b   = (const float*)d_in[7];
  const float* dec1_Wk  = (const float*)d_in[8];
  const float* dec1_Wr  = (const float*)d_in[9];
  const float* dec1_b   = (const float*)d_in[10];
  const float* dec2_Wk  = (const float*)d_in[11];
  const float* dec2_Wr  = (const float*)d_in[12];
  const float* dec2_b   = (const float*)d_in[13];
  const float* fc_W     = (const float*)d_in[14];
  const float* fc_b     = (const float*)d_in[15];
  float* out            = (float*)d_out;

  // workspace carve-up (256B aligned)
  size_t off = 0;
  auto carve = [&](size_t bytes) -> char* {
    char* p = (char*)d_ws + off;
    off += (bytes + 255) & ~(size_t)255;
    return p;
  };
  bf16*  inp_bf = (bf16*)carve((long)B_ * TPAST * FIN * 2);
  bf16*  tgt_bf = (bf16*)carve(MD * 32 * 2);
  bf16*  WtE1   = (bf16*)carve((long)G4H * (FIN + H) * 2);  // [2048][576]
  bf16*  WtE2   = (bf16*)carve((long)G4H * (2 * H) * 2);    // [2048][1024]
  bf16*  Wd1k   = (bf16*)carve((long)G4H * 32 * 2);         // [2048][32]
  bf16*  Wd1rT  = (bf16*)carve((long)G4H * H * 2);          // [2048][512]
  bf16*  WtD2   = (bf16*)carve((long)G4H * H * 2);          // [2048][512]
  bf16*  WfcT   = (bf16*)carve((long)FOUT * H * 2);         // [16][512]
  bf16*  hA     = (bf16*)carve((long)B_ * H * 2);
  bf16*  hB     = (bf16*)carve((long)B_ * H * 2);
  float* cstate = (float*)carve((long)B_ * H * 4);
  bf16*  h1s    = (bf16*)carve((long)TPAST * B_ * H * 2);   // [T][B][H]
  float* r2buf  = (float*)carve((long)B_ * G4H * 4);
  bf16*  h1d    = (bf16*)carve(MD * H * 2);
  float* c1d    = (float*)carve(MD * H * 4);
  bf16*  h2d    = (bf16*)carve(MD * H * 2);

  const int TB = 256;

  // ---- prep ----
  {
    long n = (long)B_ * TPAST * FIN;
    k_cvt_f32_bf16<<<ceil_div(n, TB), TB, 0, stream>>>(inp, inp_bf, n);
  }
  k_prep_tgt<<<ceil_div(MD * 32, TB), TB, 0, stream>>>(tgt, tgt_bf);
  k_prep_cat<<<ceil_div((long)G4H * (FIN + H), TB), TB, 0, stream>>>(
      enc1_Wk, FIN, enc1_Wr, H, G4H, WtE1);
  k_prep_cat<<<ceil_div((long)G4H * (2 * H), TB), TB, 0, stream>>>(
      enc2_Wk, H, enc2_Wr, H, G4H, WtE2);
  k_prep_pad<<<ceil_div(2048L * 32, TB), TB, 0, stream>>>(dec1_Wk, Wd1k);
  k_prep_cat<<<ceil_div((long)G4H * H, TB), TB, 0, stream>>>(
      dec1_Wr, H, nullptr, 0, G4H, Wd1rT);
  k_prep_sum<<<ceil_div((long)G4H * H, TB), TB, 0, stream>>>(
      dec2_Wk, dec2_Wr, H, G4H, WtD2);
  k_prep_cat<<<ceil_div((long)FOUT * H, TB), TB, 0, stream>>>(
      fc_W, H, nullptr, 0, FOUT, WfcT);
  k_zero_bf16<<<ceil_div((long)B_ * H, TB), TB, 0, stream>>>(hA, (long)B_ * H);
  k_zero_f32<<<ceil_div((long)B_ * H, TB), TB, 0, stream>>>(cstate, (long)B_ * H);

  // ---- encoder layer 1 (serial over T) ----
  // blocks = (rows/256 row groups) * 32 col tiles = 2*32 = 64
  bf16* hc = hA;
  bf16* hn = hB;
  for (int t = 0; t < TPAST; ++t) {
    k_lstm_step<<<64, TB, 0, stream>>>(
        inp_bf + (long)t * FIN, (long)TPAST * FIN, FIN, hc, WtE1, FIN + H,
        enc1_b, cstate, hn, h1s + (long)t * B_ * H, B_);
    bf16* tmp = hc; hc = hn; hn = tmp;
  }
  // hc now holds h1 final; cstate holds c1 final (layer2 initial state)

  // ---- encoder layer 2 (serial over T, reads h1s) ----
  for (int t = 0; t < TPAST; ++t) {
    k_lstm_step<<<64, TB, 0, stream>>>(
        h1s + (long)t * B_ * H, (long)H, H, hc, WtE2, 2 * H,
        enc2_b, cstate, hn, nullptr, B_);
    bf16* tmp = hc; hc = hn; hn = tmp;
  }
  // hc == h2 (bf16), cstate == c2

  // ---- decoder (parallel over B*TFUT) ----
  // k_r2: (512/32) x 128 tiles = 2048 -> 256 blocks
  k_r2<<<256, TB, 0, stream>>>(hc, Wd1rT, r2buf);
  k_dec1<<<6144, TB, 0, stream>>>(tgt_bf, Wd1k, r2buf, cstate, dec1_b, h1d, c1d);
  // decoder layer 2: rows=24576 -> (24576/256 groups)*32 col tiles = 3072 blocks
  k_lstm_step<<<3072, TB, 0, stream>>>(
      h1d, (long)H, H, nullptr, WtD2, H, dec2_b, c1d, h2d, nullptr, (int)MD);
  // ---- output projection: 24576/32 = 768 row blocks -> 96 blocks ----
  k_fc<<<96, TB, 0, stream>>>(h2d, WfcT, fc_b, out);
}
